// MaskRCNNModel_2207613190478
// MI455X (gfx1250) — matrix operand order repaired and checked by reading, not compiled, not run on hardware
//
#include <hip/hip_runtime.h>
#include <hip/hip_bf16.h>

#define IMG_H 800
#define IMG_W 1216
#define FH 200
#define FW 304
#define FHW (FH*FW)
#define NA 3
#define NANCH (NA*FHW)
#define PRE_NMS 1000
#define POST_NMS 100
#define NMS_THRESH 0.7f
#define NUM_CLASSES 81

typedef __attribute__((ext_vector_type(16))) __bf16 v16bf;
typedef __attribute__((ext_vector_type(8)))  float  v8f;

static __device__ __forceinline__ __bf16 f2bf(float f) {
  __hip_bfloat16 t = __float2bfloat16(f);
  __bf16 r;
  __builtin_memcpy(&r, &t, 2);
  return r;
}
static __device__ __forceinline__ int iclamp(int v, int lo, int hi) {
  return v < lo ? lo : (v > hi ? hi : v);
}

// ---------------------------------------------------------------------------
// Generic implicit-GEMM convolution on WMMA bf16 tensor pipes.
//   out[b, n, oy, ox] = act( sum_{ci,fy,fx} in[b,ci,oy*s-p+fy,ox*s-p+fx] * w[n,ci,fy,fx] )
// M = B*OH*OW (rows, im2col), N = Cout, K = Cin*KH*KW.
// Workgroup: 128 threads (4 waves) -> 64x64 tile; K-stage depth 64
// (2 WMMA k-substeps per LDS fill -> 8 v_wmma per wave per stage).
// K1X1 specialization: 1x1/stride1/pad0 (FC layers, heads) -> no filter
// decomposition, no pad test in the gather.
// ---------------------------------------------------------------------------
template <bool K1X1>
__global__ __launch_bounds__(128)
void conv_wmma(const float* __restrict__ in, const float* __restrict__ wgt,
               float* __restrict__ out,
               int B, int Cin, int H, int W, int Cout, int KH, int KW,
               int stride, int pad, int OH, int OW, int relu)
{
  const int OHW = OH * OW;
  const int KHW = KH * KW;
  const int M = B * OHW;
  const int N = Cout;
  const int K = K1X1 ? Cin : Cin * KHW;
  const int mBase = blockIdx.y * 64;
  const int nBase = blockIdx.x * 64;

  __shared__ __bf16 lA[64 * 66];   // [row(M-tile)][k 0..63]  padded stride
  __shared__ __bf16 lB[64 * 66];   // [k 0..63][col(N-tile)]  padded stride

  const int tid  = threadIdx.x;
  const int lane = tid & 31;
  const int wave = tid >> 5;
  const int wy   = wave >> 1;      // 0..1 : which 32-row half
  const int wx   = wave & 1;       // 0..1 : which 32-col half
  const int half = lane >> 4;      // 0..1
  const int r16  = lane & 15;      // 0..15

  v8f acc[2][2];
#pragma unroll
  for (int i = 0; i < 2; ++i)
#pragma unroll
    for (int j = 0; j < 2; ++j)
#pragma unroll
      for (int e = 0; e < 8; ++e) acc[i][j][e] = 0.0f;

  const __bf16 bz = f2bf(0.0f);
  const int KT = (K + 63) / 64;

  for (int kt = 0; kt < KT; ++kt) {
    const int kBase = kt * 64;

    // gfx1250 prefetch: pull the next K-stage of the weight stream toward
    // this WGP while the current stage is being staged/consumed.
    {
      int pk = kBase + 64 + (tid >> 2);        // next-stage k row (32 rows/WG)
      int pn = nBase + ((tid & 3) << 4);       // spread across the N-tile
      if (pk < K && pn < N) {
        const float* pf = K1X1 ? &wgt[pn * Cin + pk]
                               : &wgt[(size_t)pn * Cin * KHW + pk];
        __builtin_prefetch(pf, 0, 1);
      }
    }

    // ---- stage A tile: 64 rows x 64 k (im2col gather, f32 -> bf16) ----
#pragma unroll
    for (int t = 0; t < 32; ++t) {
      int idx = t * 128 + tid;              // 4096 elements
      int row = idx >> 6, col = idx & 63;
      int m = mBase + row, k = kBase + col;
      __bf16 v = bz;
      if (m < M && k < K) {
        int b  = m / OHW;
        int rm = m - b * OHW;
        int oy = rm / OW, ox = rm - oy * OW;
        if (K1X1) {
          v = f2bf(in[((b * Cin + k) * H + oy) * W + ox]);
        } else {
          int ci = k / KHW;
          int rk = k - ci * KHW;
          int fy = rk / KW, fx = rk - fy * KW;
          int iy = oy * stride - pad + fy;
          int ix = ox * stride - pad + fx;
          if (iy >= 0 && iy < H && ix >= 0 && ix < W)
            v = f2bf(in[((b * Cin + ci) * H + iy) * W + ix]);
        }
      }
      lA[row * 66 + col] = v;
    }
    // ---- stage B tile: 64 k x 64 cols (OIHW weights, f32 -> bf16) ----
#pragma unroll
    for (int t = 0; t < 32; ++t) {
      int idx = t * 128 + tid;
      int kk = idx >> 6, nn = idx & 63;
      int k = kBase + kk, n = nBase + nn;
      __bf16 v = bz;
      if (k < K && n < N) {
        if (K1X1) {
          v = f2bf(wgt[n * Cin + k]);
        } else {
          int ci = k / KHW;
          int rk = k - ci * KHW;
          int fy = rk / KW, fx = rk - fy * KW;
          v = f2bf(wgt[((n * Cin + ci) * KH + fy) * KW + fx]);
        }
      }
      lB[kk * 66 + nn] = v;
    }
    __syncthreads();

    // ---- 2 k-substeps of 32: fragments per ISA 16-bit A(16x32)/B(32x16) ----
#pragma unroll
    for (int ks = 0; ks < 64; ks += 32) {
      v16bf fa[2], fb[2];
#pragma unroll
      for (int i = 0; i < 2; ++i) {
        int ar = wy * 32 + i * 16 + r16;
#pragma unroll
        for (int e = 0; e < 8; ++e) fa[i][e]     = lA[ar * 66 + ks + half * 8 + e];
#pragma unroll
        for (int e = 0; e < 8; ++e) fa[i][8 + e] = lA[ar * 66 + ks + 16 + half * 8 + e];
      }
#pragma unroll
      for (int j = 0; j < 2; ++j) {
        int bc = wx * 32 + j * 16 + r16;
#pragma unroll
        for (int e = 0; e < 16; ++e) fb[j][e] = lB[(ks + (e >> 3) * 16 + half * 8 + (e & 7)) * 66 + bc];
      }
#pragma unroll
      for (int i = 0; i < 2; ++i)
#pragma unroll
        for (int j = 0; j < 2; ++j)
          acc[i][j] = __builtin_amdgcn_wmma_f32_16x16x32_bf16(
              false, fa[i], false, fb[j], (short)0, acc[i][j], false, false);
    }
    __syncthreads();
  }

  // ---- epilogue: scatter C (f32 16x16 VGPR layout), optional ReLU ----
#pragma unroll
  for (int i = 0; i < 2; ++i)
#pragma unroll
    for (int j = 0; j < 2; ++j)
#pragma unroll
      for (int rr = 0; rr < 8; ++rr) {
        int gm = mBase + wy * 32 + i * 16 + half * 8 + rr;
        int gn = nBase + wx * 32 + j * 16 + r16;
        if (gm < M && gn < N) {
          float v = acc[i][j][rr];
          if (relu) v = fmaxf(v, 0.0f);
          int b  = gm / OHW;
          int rm = gm - b * OHW;
          int oy = rm / OW, ox = rm - oy * OW;
          out[((b * Cout + gn) * OH + oy) * OW + ox] = v;
        }
      }
}

// ---------------------------------------------------------------------------
// Anchor decode
// ---------------------------------------------------------------------------
__global__ void decode_kernel(const float* __restrict__ deltas, float* __restrict__ boxes)
{
  int i = blockIdx.x * blockDim.x + threadIdx.x;
  if (i >= NANCH) return;
  int a = i / FHW;
  int r = i - a * FHW;
  int y = r / FW, x = r - y * FW;
  float sz  = (a == 0) ? 32.f : ((a == 1) ? 64.f : 128.f);
  float cxa = (x + 0.5f) * 4.f, cya = (y + 0.5f) * 4.f;
  float dx = deltas[((a * 4 + 0) * FH + y) * FW + x];
  float dy = deltas[((a * 4 + 1) * FH + y) * FW + x];
  float dw = deltas[((a * 4 + 2) * FH + y) * FW + x];
  float dh = deltas[((a * 4 + 3) * FH + y) * FW + x];
  const float clipv = 4.135166556742356f;  // log(1000/16)
  dw = fminf(dw, clipv);
  dh = fminf(dh, clipv);
  float cx = dx * sz + cxa, cy = dy * sz + cya;
  float w = __expf(dw) * sz, h = __expf(dh) * sz;
  float x1 = cx - 0.5f * w, y1 = cy - 0.5f * h;
  float x2 = cx + 0.5f * w, y2 = cy + 0.5f * h;
  boxes[i * 4 + 0] = fminf(fmaxf(x1, 0.f), IMG_W - 1.f);
  boxes[i * 4 + 1] = fminf(fmaxf(y1, 0.f), IMG_H - 1.f);
  boxes[i * 4 + 2] = fminf(fmaxf(x2, 0.f), IMG_W - 1.f);
  boxes[i * 4 + 3] = fminf(fmaxf(y2, 0.f), IMG_H - 1.f);
}

// ---------------------------------------------------------------------------
// Top-K (radix-histogram select): meta = [thrBin, cGreater, cntHi, cntEq]
// ---------------------------------------------------------------------------
static __device__ __forceinline__ unsigned fkey(float s) {
  unsigned u = __float_as_uint(s);
  return (u & 0x80000000u) ? ~u : (u | 0x80000000u);
}

__global__ void topk_init(int* hist, int* meta, float* tsc, float* tbx)
{
  int i = blockIdx.x * blockDim.x + threadIdx.x;
  if (i < 2048) hist[i] = 0;
  if (i < 8) meta[i] = 0;
  if (i < PRE_NMS) {
    tsc[i] = -3.0e38f;
    tbx[i * 4 + 0] = 0.f; tbx[i * 4 + 1] = 0.f;
    tbx[i * 4 + 2] = 0.f; tbx[i * 4 + 3] = 0.f;
  }
}

__global__ void topk_hist(const float* __restrict__ obj, int* __restrict__ hist)
{
  __shared__ int lh[2048];
  for (int i = threadIdx.x; i < 2048; i += blockDim.x) lh[i] = 0;
  __syncthreads();
  for (int i = blockIdx.x * blockDim.x + threadIdx.x; i < NANCH; i += gridDim.x * blockDim.x)
    atomicAdd(&lh[fkey(obj[i]) >> 21], 1);
  __syncthreads();
  for (int i = threadIdx.x; i < 2048; i += blockDim.x)
    if (lh[i]) atomicAdd(&hist[i], lh[i]);
}

__global__ void topk_thresh(const int* __restrict__ hist, int* __restrict__ meta)
{
  if (threadIdx.x == 0 && blockIdx.x == 0) {
    int cum = 0, thr = 0;
    for (int b = 2047; b >= 0; --b) {
      int c = hist[b];
      if (cum + c >= PRE_NMS) { thr = b; break; }
      cum += c;
    }
    meta[0] = thr;
    meta[1] = cum;
  }
}

__global__ void topk_collect_hi(const float* __restrict__ obj, const float* __restrict__ boxes,
                                int* __restrict__ meta, float* __restrict__ tsc, float* __restrict__ tbx)
{
  int i = blockIdx.x * blockDim.x + threadIdx.x;
  if (i >= NANCH) return;
  float s = obj[i];
  if ((int)(fkey(s) >> 21) > meta[0]) {
    int pos = atomicAdd(&meta[2], 1);
    if (pos < PRE_NMS) {
      tsc[pos] = s;
      tbx[pos * 4 + 0] = boxes[i * 4 + 0];
      tbx[pos * 4 + 1] = boxes[i * 4 + 1];
      tbx[pos * 4 + 2] = boxes[i * 4 + 2];
      tbx[pos * 4 + 3] = boxes[i * 4 + 3];
    }
  }
}

__global__ void topk_collect_eq(const float* __restrict__ obj, const float* __restrict__ boxes,
                                int* __restrict__ meta, float* __restrict__ tsc, float* __restrict__ tbx)
{
  int i = blockIdx.x * blockDim.x + threadIdx.x;
  if (i >= NANCH) return;
  float s = obj[i];
  if ((int)(fkey(s) >> 21) == meta[0]) {
    int pos  = atomicAdd(&meta[3], 1);
    int slot = meta[1] + pos;
    if (slot < PRE_NMS) {
      tsc[slot] = s;
      tbx[slot * 4 + 0] = boxes[i * 4 + 0];
      tbx[slot * 4 + 1] = boxes[i * 4 + 1];
      tbx[slot * 4 + 2] = boxes[i * 4 + 2];
      tbx[slot * 4 + 3] = boxes[i * 4 + 3];
    }
  }
}

// ---------------------------------------------------------------------------
// NMS: single workgroup, everything in LDS (fits easily in 320KB WGP LDS)
// ---------------------------------------------------------------------------
__global__ __launch_bounds__(1024)
void nms_kernel(const float* __restrict__ tsc, const float* __restrict__ tbx,
                float* __restrict__ rois, int* __restrict__ valid)
{
  __shared__ float sc[PRE_NMS];
  __shared__ float bx[PRE_NMS * 4];
  __shared__ float rv[1024];
  __shared__ int   ri[1024];
  __shared__ int   skeep[POST_NMS];
  __shared__ int   svalid[POST_NMS];

  const int t = threadIdx.x;
  if (t < PRE_NMS) {
    sc[t] = tsc[t];
    bx[t * 4 + 0] = tbx[t * 4 + 0];
    bx[t * 4 + 1] = tbx[t * 4 + 1];
    bx[t * 4 + 2] = tbx[t * 4 + 2];
    bx[t * 4 + 3] = tbx[t * 4 + 3];
  }
  __syncthreads();

  for (int it = 0; it < POST_NMS; ++it) {
    rv[t] = (t < PRE_NMS) ? sc[t] : -3.0e38f;
    ri[t] = t;
    __syncthreads();
    for (int s = 512; s > 0; s >>= 1) {
      if (t < s && rv[t + s] > rv[t]) { rv[t] = rv[t + s]; ri[t] = ri[t + s]; }
      __syncthreads();
    }
    int   b0 = ri[0];
    int   ok = rv[0] > -1.0e38f;
    float bx1 = bx[b0 * 4 + 0], by1 = bx[b0 * 4 + 1];
    float bx2 = bx[b0 * 4 + 2], by2 = bx[b0 * 4 + 3];
    if (t == 0) { skeep[it] = b0; svalid[it] = ok; }
    __syncthreads();
    if (t < PRE_NMS) {
      if (!ok) {
        sc[t] = -3.0e38f;
      } else {
        float ax1 = bx[t * 4 + 0], ay1 = bx[t * 4 + 1];
        float ax2 = bx[t * 4 + 2], ay2 = bx[t * 4 + 3];
        float areaA = (ax2 - ax1) * (ay2 - ay1);
        float areaB = (bx2 - bx1) * (by2 - by1);
        float ix1 = fmaxf(ax1, bx1), iy1 = fmaxf(ay1, by1);
        float ix2 = fminf(ax2, bx2), iy2 = fminf(ay2, by2);
        float iw = fmaxf(ix2 - ix1, 0.f), ih = fmaxf(iy2 - iy1, 0.f);
        float inter = iw * ih;
        float iou = inter / (areaA + areaB - inter + 1e-6f);
        if (iou > NMS_THRESH) sc[t] = -3.0e38f;
      }
    }
    __syncthreads();
  }

  if (t < POST_NMS) {
    int   k  = skeep[t];
    float vf = svalid[t] ? 1.f : 0.f;
    rois[t * 4 + 0] = bx[k * 4 + 0] * vf;
    rois[t * 4 + 1] = bx[k * 4 + 1] * vf;
    rois[t * 4 + 2] = bx[k * 4 + 2] * vf;
    rois[t * 4 + 3] = bx[k * 4 + 3] * vf;
    valid[t] = svalid[t];
  }
}

// ---------------------------------------------------------------------------
// ROI-align (P in {7,14}), 2x2 subsamples averaged, bilinear taps
// ---------------------------------------------------------------------------
__global__ void roi_align_kernel(const float* __restrict__ feat, const float* __restrict__ rois,
                                 float* __restrict__ out, int P)
{
  int idx = blockIdx.x * blockDim.x + threadIdx.x;
  int total = POST_NMS * 256 * P * P;
  if (idx >= total) return;
  int px = idx % P; int tmp = idx / P;
  int py = tmp % P; tmp /= P;
  int c  = tmp % 256;
  int r  = tmp / 256;

  float x1 = rois[r * 4 + 0] * 0.25f, y1 = rois[r * 4 + 1] * 0.25f;
  float x2 = rois[r * 4 + 2] * 0.25f, y2 = rois[r * 4 + 3] * 0.25f;
  float h = fmaxf(y2 - y1, 1.f), w = fmaxf(x2 - x1, 1.f);
  float invS = 1.f / (float)(2 * P);
  const float* f = feat + (size_t)c * FH * FW;

  float acc = 0.f;
#pragma unroll
  for (int sy = 0; sy < 2; ++sy)
#pragma unroll
    for (int sx = 0; sx < 2; ++sx) {
      float ys = y1 + (2 * py + sy + 0.5f) * h * invS;
      float xs = x1 + (2 * px + sx + 0.5f) * w * invS;
      float y0 = floorf(ys), x0 = floorf(xs);
      float ly = ys - y0, lx = xs - x0;
      int y0i = iclamp((int)y0, 0, FH - 1);
      int y1i = iclamp((int)y0 + 1, 0, FH - 1);
      int x0i = iclamp((int)x0, 0, FW - 1);
      int x1i = iclamp((int)x0 + 1, 0, FW - 1);
      float v00 = f[y0i * FW + x0i], v01 = f[y0i * FW + x1i];
      float v10 = f[y1i * FW + x0i], v11 = f[y1i * FW + x1i];
      acc += v00 * (1.f - ly) * (1.f - lx) + v01 * (1.f - ly) * lx
           + v10 * ly * (1.f - lx)        + v11 * ly * lx;
    }
  out[idx] = acc * 0.25f;
}

// ---------------------------------------------------------------------------
// Softmax head -> scores/labels
// ---------------------------------------------------------------------------
__global__ void cls_head_kernel(const float* __restrict__ logits, const int* __restrict__ valid,
                                float* __restrict__ scores, int* __restrict__ labels)
{
  int r = blockIdx.x * blockDim.x + threadIdx.x;
  if (r >= POST_NMS) return;
  const float* l = logits + r * NUM_CLASSES;
  float m = l[0];
  for (int c = 1; c < NUM_CLASSES; ++c) m = fmaxf(m, l[c]);
  float sum = 0.f;
  for (int c = 0; c < NUM_CLASSES; ++c) sum += __expf(l[c] - m);
  float inv = 1.f / sum;
  float best = -1.f; int bi = 1;
  for (int c = 1; c < NUM_CLASSES; ++c) {
    float p = __expf(l[c] - m) * inv;
    if (p > best) { best = p; bi = c; }
  }
  int v = valid[r];
  scores[r] = best * (v ? 1.f : 0.f);
  labels[r] = v ? bi : 0;
}

// ---------------------------------------------------------------------------
// Deconv 2x2 stride2 == 4 independent 1x1 convs: weight repack + scatter
// ---------------------------------------------------------------------------
__global__ void repack_deconv_w(const float* __restrict__ wd, float* __restrict__ wr)
{
  int i = blockIdx.x * blockDim.x + threadIdx.x;
  if (i >= 1024 * 256) return;
  int ci = i % 256;
  int n  = i / 256;                 // n = co*4 + di*2 + dj
  int dj = n & 1, di = (n >> 1) & 1, co = n >> 2;
  wr[n * 256 + ci] = wd[((ci * 256 + co) * 2 + di) * 2 + dj];
}

__global__ void deconv_scatter(const float* __restrict__ dect, float* __restrict__ outp)
{
  int idx = blockIdx.x * blockDim.x + threadIdx.x;
  const int total = POST_NMS * 1024 * 196;
  if (idx >= total) return;
  int j = idx % 14; int t = idx / 14;
  int i2 = t % 14;  t /= 14;
  int n = t % 1024;
  int r = t / 1024;
  int dj = n & 1, di = (n >> 1) & 1, co = n >> 2;
  outp[((r * 256 + co) * 28 + (2 * i2 + di)) * 28 + (2 * j + dj)] = fmaxf(dect[idx], 0.f);
}

// ---------------------------------------------------------------------------
// Mask select + sigmoid; output packing
// ---------------------------------------------------------------------------
__global__ void mask_out_kernel(const float* __restrict__ mpred, const int* __restrict__ labels,
                                const int* __restrict__ valid, float* __restrict__ outmask)
{
  int idx = blockIdx.x * blockDim.x + threadIdx.x;
  const int total = POST_NMS * 784;
  if (idx >= total) return;
  int p = idx % 784;
  int r = idx / 784;
  int lab = labels[r];
  float v = mpred[((size_t)r * NUM_CLASSES + lab) * 784 + p];
  float s = 1.f / (1.f + __expf(-v));
  outmask[idx] = s * (valid[r] ? 1.f : 0.f);
}

__global__ void pack_kernel(const float* __restrict__ rois, const int* __restrict__ labels,
                            const float* __restrict__ scores, float* __restrict__ out)
{
  int t = blockIdx.x * blockDim.x + threadIdx.x;
  if (t < 400) out[t] = rois[t];
  if (t < POST_NMS) {
    out[400 + t]   = (float)labels[t];
    out[78900 + t] = scores[t];
  }
}

// ---------------------------------------------------------------------------
extern "C" void kernel_launch(void* const* d_in, const int* in_sizes, int n_in,
                              void* d_out, int out_size, void* d_ws, size_t ws_size,
                              hipStream_t stream)
{
  (void)in_sizes; (void)n_in; (void)out_size; (void)ws_size;

  const float* image        = (const float*)d_in[0];
  const float* stem_w1      = (const float*)d_in[1];
  const float* stem_w2      = (const float*)d_in[2];
  const float* rpn_w        = (const float*)d_in[3];
  const float* rpn_cls_w    = (const float*)d_in[4];
  const float* rpn_box_w    = (const float*)d_in[5];
  const float* fc1_w        = (const float*)d_in[6];
  const float* fc2_w        = (const float*)d_in[7];
  const float* cls_w        = (const float*)d_in[8];
  const float* mask_w1      = (const float*)d_in[9];
  const float* mask_w2      = (const float*)d_in[10];
  const float* mask_w3      = (const float*)d_in[11];
  const float* mask_w4      = (const float*)d_in[12];
  const float* mask_deconvw = (const float*)d_in[13];
  const float* mask_pred_w  = (const float*)d_in[14];
  float* out = (float*)d_out;

  // bump allocator over workspace (256B aligned slices)
  char*  base = (char*)d_ws;
  size_t off  = 0;
  auto alloc = [&](size_t elems) -> float* {
    float* p = (float*)(base + off);
    off += ((elems * sizeof(float) + 255) / 256) * 256;
    return p;
  };

  float* stem1    = alloc((size_t)64 * FHW);
  float* feat     = alloc((size_t)256 * FHW);
  float* rpnT     = alloc((size_t)256 * FHW);
  float* obj      = alloc((size_t)3 * FHW);
  float* deltas   = alloc((size_t)12 * FHW);
  float* boxes    = alloc((size_t)NANCH * 4);
  int*   hist     = (int*)alloc(2048);
  int*   meta     = (int*)alloc(8);
  float* tsc      = alloc(PRE_NMS);
  float* tbx      = alloc((size_t)PRE_NMS * 4);
  float* rois     = alloc((size_t)POST_NMS * 4);
  int*   valid    = (int*)alloc(POST_NMS);
  int*   labels   = (int*)alloc(POST_NMS);
  float* scores   = alloc(POST_NMS);
  float* pooled7  = alloc((size_t)POST_NMS * 256 * 49);
  float* fc1o     = alloc((size_t)POST_NMS * 1024);
  float* fc2o     = alloc((size_t)POST_NMS * 1024);
  float* logits   = alloc((size_t)POST_NMS * NUM_CLASSES);
  float* pooled14 = alloc((size_t)POST_NMS * 256 * 196);
  float* mA       = alloc((size_t)POST_NMS * 256 * 196);
  float* mB       = alloc((size_t)POST_NMS * 256 * 196);
  float* wr       = alloc((size_t)1024 * 256);
  float* dect     = alloc((size_t)POST_NMS * 1024 * 196);
  float* dcout    = alloc((size_t)POST_NMS * 256 * 784);
  float* mpred    = alloc((size_t)POST_NMS * NUM_CLASSES * 784);

  auto launch_conv = [&](const float* in, const float* wgt, float* o,
                         int B, int Cin, int H, int W, int Cout, int KH, int KW,
                         int stride, int pad, int OH, int OW, int relu) {
    int M = B * OH * OW;
    dim3 g((Cout + 63) / 64, (M + 63) / 64);
    if (KH == 1 && KW == 1 && stride == 1 && pad == 0)
      conv_wmma<true><<<g, 128, 0, stream>>>(in, wgt, o, B, Cin, H, W, Cout, KH, KW,
                                             stride, pad, OH, OW, relu);
    else
      conv_wmma<false><<<g, 128, 0, stream>>>(in, wgt, o, B, Cin, H, W, Cout, KH, KW,
                                              stride, pad, OH, OW, relu);
  };

  // ---- backbone + RPN (WMMA implicit-GEMM convs) ----
  launch_conv(image, stem_w1, stem1, 1, 3,   IMG_H, IMG_W, 64,  7, 7, 4, 1, FH, FW, 1);
  launch_conv(stem1, stem_w2, feat,  1, 64,  FH, FW,       256, 3, 3, 1, 1, FH, FW, 1);
  launch_conv(feat,  rpn_w,   rpnT,  1, 256, FH, FW,       256, 3, 3, 1, 1, FH, FW, 1);
  launch_conv(rpnT, rpn_cls_w, obj,    1, 256, FH, FW, 3,  1, 1, 1, 0, FH, FW, 0);
  launch_conv(rpnT, rpn_box_w, deltas, 1, 256, FH, FW, 12, 1, 1, 1, 0, FH, FW, 0);

  // ---- decode + radix-select top-1000 + NMS ----
  decode_kernel<<<(NANCH + 255) / 256, 256, 0, stream>>>(deltas, boxes);
  topk_init<<<8, 256, 0, stream>>>(hist, meta, tsc, tbx);
  topk_hist<<<256, 256, 0, stream>>>(obj, hist);
  topk_thresh<<<1, 1, 0, stream>>>(hist, meta);
  topk_collect_hi<<<(NANCH + 255) / 256, 256, 0, stream>>>(obj, boxes, meta, tsc, tbx);
  topk_collect_eq<<<(NANCH + 255) / 256, 256, 0, stream>>>(obj, boxes, meta, tsc, tbx);
  nms_kernel<<<1, 1024, 0, stream>>>(tsc, tbx, rois, valid);

  // ---- box head: ROI-align 7x7 -> FC(WMMA) -> softmax ----
  roi_align_kernel<<<(POST_NMS * 256 * 49 + 255) / 256, 256, 0, stream>>>(feat, rois, pooled7, 7);
  launch_conv(pooled7, fc1_w, fc1o, POST_NMS, 256 * 49, 1, 1, 1024,        1, 1, 1, 0, 1, 1, 1);
  launch_conv(fc1o,    fc2_w, fc2o, POST_NMS, 1024,     1, 1, 1024,        1, 1, 1, 0, 1, 1, 1);
  launch_conv(fc2o,    cls_w, logits, POST_NMS, 1024,   1, 1, NUM_CLASSES, 1, 1, 1, 0, 1, 1, 0);
  cls_head_kernel<<<1, 128, 0, stream>>>(logits, valid, scores, labels);

  // ---- mask head: ROI-align 14x14 -> 4 convs -> deconv(as GEMM) -> pred ----
  roi_align_kernel<<<(POST_NMS * 256 * 196 + 255) / 256, 256, 0, stream>>>(feat, rois, pooled14, 14);
  launch_conv(pooled14, mask_w1, mA, POST_NMS, 256, 14, 14, 256, 3, 3, 1, 1, 14, 14, 1);
  launch_conv(mA,       mask_w2, mB, POST_NMS, 256, 14, 14, 256, 3, 3, 1, 1, 14, 14, 1);
  launch_conv(mB,       mask_w3, mA, POST_NMS, 256, 14, 14, 256, 3, 3, 1, 1, 14, 14, 1);
  launch_conv(mA,       mask_w4, mB, POST_NMS, 256, 14, 14, 256, 3, 3, 1, 1, 14, 14, 1);
  repack_deconv_w<<<(1024 * 256 + 255) / 256, 256, 0, stream>>>(mask_deconvw, wr);
  launch_conv(mB, wr, dect, POST_NMS, 256, 14, 14, 1024, 1, 1, 1, 0, 14, 14, 0);
  deconv_scatter<<<(POST_NMS * 1024 * 196 + 255) / 256, 256, 0, stream>>>(dect, dcout);
  launch_conv(dcout, mask_pred_w, mpred, POST_NMS, 256, 28, 28, NUM_CLASSES, 1, 1, 1, 0, 28, 28, 0);

  // ---- outputs: [rois(400) | labels(100) | masks(78400) | scores(100)] ----
  mask_out_kernel<<<(POST_NMS * 784 + 255) / 256, 256, 0, stream>>>(mpred, labels, valid, out + 500);
  pack_kernel<<<2, 256, 0, stream>>>(rois, labels, scores, out);
}
